// SparseAutoencoder_4071628997287
// MI455X (gfx1250) — matrix operand order repaired and checked by reading, not compiled, not run on hardware
//
#include <hip/hip_runtime.h>
#include <stdint.h>

// ---------------------------------------------------------------------------
// Types for CDNA5 WMMA (wave32): v_wmma_f32_16x16x32_bf16
// ---------------------------------------------------------------------------
typedef __attribute__((ext_vector_type(16))) __bf16 v16bf;
typedef __attribute__((ext_vector_type(8)))  __bf16 v8bf;
typedef __attribute__((ext_vector_type(4)))  __bf16 v4bf;
typedef __attribute__((ext_vector_type(8)))  float  v8f;

union ABFrag { v16bf v; v8bf h[2]; };

// ---------------------------------------------------------------------------
// NT GEMM: C[M,N] = A[M,K] * B[N,K]^T   (both row-major, contraction over
// the contiguous dim of both), optional fused bias[n] + ReLU epilogue.
// f32 in HBM -> cvt_pk_bf16 while staging through LDS -> f32 accumulate.
// Block tile 128x128, 8 waves, wave tile 32x64 (2x4 WMMA tiles), BK = 32.
// Double-buffered LDS: one barrier per K-step; global loads for stage k+1
// overlap the 8 WMMAs of stage k; global_prefetch warms stage k+2.
// All problem dims are multiples of the tiles -> no edge handling.
// ---------------------------------------------------------------------------
template<bool FUSE_BIAS_RELU>
__global__ __launch_bounds__(256)
void gemm_nt_wmma(const float* __restrict__ A, const float* __restrict__ B,
                  const float* __restrict__ bias, float* __restrict__ C,
                  int M, int N, int Ksz)
{
    constexpr int BM = 128, BN = 128, BK = 32;
    constexpr int PAD = 8;                      // 80B row stride in LDS
    __shared__ __bf16 sA[2][BM][BK + PAD];
    __shared__ __bf16 sB[2][BN][BK + PAD];

    const int t    = threadIdx.x;
    const int m0   = blockIdx.y * BM;
    const int n0   = blockIdx.x * BN;
    const int wave = t >> 5;
    const int lane = t & 31;
    const int l15  = lane & 15;
    const int hi   = lane >> 4;                 // which 16-lane half
    const int wr   = wave >> 1;                 // 0..3 : 32-row group
    const int wc   = wave & 1;                  // 0..1 : 64-col group

    const float* Abase = A + (size_t)m0 * Ksz;
    const float* Bbase = B + (size_t)n0 * Ksz;

    // Per-thread staging slots: 4 float4 per matrix per stage.
    const int idx4_0 = t;                       // + i*256, i in 0..3
    // row = idx4 >> 3 (8 float4 per 32-wide row), c4 = idx4 & 7

    auto issueLoads = [&](int k0, float4 ra[4], float4 rb[4]) {
        #pragma unroll
        for (int i = 0; i < 4; ++i) {
            const int idx4 = idx4_0 + i * 256;
            const int row  = idx4 >> 3;
            const int c4   = idx4 & 7;
            ra[i] = *(const float4*)(Abase + (size_t)row * Ksz + k0 + c4 * 4);
            rb[i] = *(const float4*)(Bbase + (size_t)row * Ksz + k0 + c4 * 4);
        }
    };
    auto storeStage = [&](int buf, const float4 ra[4], const float4 rb[4]) {
        #pragma unroll
        for (int i = 0; i < 4; ++i) {
            const int idx4 = idx4_0 + i * 256;
            const int row  = idx4 >> 3;
            const int c4   = idx4 & 7;
            v4bf pa, pb;
            pa[0] = (__bf16)ra[i].x; pa[1] = (__bf16)ra[i].y;
            pa[2] = (__bf16)ra[i].z; pa[3] = (__bf16)ra[i].w;
            pb[0] = (__bf16)rb[i].x; pb[1] = (__bf16)rb[i].y;
            pb[2] = (__bf16)rb[i].z; pb[3] = (__bf16)rb[i].w;
            *(v4bf*)&sA[buf][row][c4 * 4] = pa;
            *(v4bf*)&sB[buf][row][c4 * 4] = pb;
        }
    };

    v8f acc[2][4];
    #pragma unroll
    for (int a = 0; a < 2; ++a)
        #pragma unroll
        for (int b = 0; b < 4; ++b)
            #pragma unroll
            for (int e = 0; e < 8; ++e) acc[a][b][e] = 0.0f;

    const int kIters = Ksz / BK;

    float4 ra[4], rb[4];
    issueLoads(0, ra, rb);
    storeStage(0, ra, rb);

    for (int kt = 0; kt < kIters; ++kt) {
        const int cur = kt & 1;
        const int nxt = cur ^ 1;
        __syncthreads();                        // sA/sB[cur] ready for all

        // Kick off next stage's global loads; they retire under the WMMAs.
        if (kt + 1 < kIters) issueLoads((kt + 1) * BK, ra, rb);
        // Warm L2 for stage k+2 (global_prefetch path, no counter cost).
        if (kt + 2 < kIters) {
            const int kp = (kt + 2) * BK;
            const int rowp = (idx4_0 >> 3);
            __builtin_prefetch(Abase + (size_t)rowp * Ksz + kp, 0, 1);
            __builtin_prefetch(Bbase + (size_t)rowp * Ksz + kp, 0, 1);
        }

        // ---- fragments per ISA 16-bit layouts ----------------------------
        // A 16x32: lanes 0-15 row M=l15 hold K{0..7,16..23}; lanes 16-31 K{8..15,24..31}
        ABFrag af[2];
        #pragma unroll
        for (int mt = 0; mt < 2; ++mt) {
            const int r = wr * 32 + mt * 16 + l15;
            af[mt].h[0] = *(const v8bf*)&sA[cur][r][hi * 8];
            af[mt].h[1] = *(const v8bf*)&sA[cur][r][hi * 8 + 16];
        }
        // B 32x16: lane l15 = column n; lanes 0-15 K{0..7,16..23}; lanes 16-31 K{8..15,24..31}
        ABFrag bfr[4];
        #pragma unroll
        for (int nt = 0; nt < 4; ++nt) {
            const int c = wc * 64 + nt * 16 + l15;
            bfr[nt].h[0] = *(const v8bf*)&sB[cur][c][hi * 8];
            bfr[nt].h[1] = *(const v8bf*)&sB[cur][c][hi * 8 + 16];
        }

        #pragma unroll
        for (int mt = 0; mt < 2; ++mt)
            #pragma unroll
            for (int nt = 0; nt < 4; ++nt)
                acc[mt][nt] = __builtin_amdgcn_wmma_f32_16x16x32_bf16(
                    false, af[mt].v, false, bfr[nt].v,
                    (short)0, acc[mt][nt], false, false);

        // Stage k+1 into the other buffer (no reader conflict; next barrier
        // publishes it).
        if (kt + 1 < kIters) storeStage(nxt, ra, rb);
    }

    // ---- epilogue: C layout -> VGPR r holds M = r + hi*8, N = l15 ---------
    #pragma unroll
    for (int nt = 0; nt < 4; ++nt) {
        const int col = n0 + wc * 64 + nt * 16 + l15;
        const float bcol = FUSE_BIAS_RELU ? bias[col] : 0.0f;
        #pragma unroll
        for (int mt = 0; mt < 2; ++mt) {
            const int rbase = m0 + wr * 32 + mt * 16 + hi * 8;
            #pragma unroll
            for (int r = 0; r < 8; ++r) {
                float v = acc[mt][nt][r];
                if (FUSE_BIAS_RELU) v = fmaxf(v + bcol, 0.0f);
                C[(size_t)(rbase + r) * N + col] = v;
            }
        }
    }
}

// ---------------------------------------------------------------------------
// Exact per-row top-K selection, in place. One 1024-thread block per row.
// Post-ReLU values are >= 0, so f32 bit patterns are monotone as uint32.
// Exact 32-bit radix select (4 x 8-bit digits, high->low). Top 16 bits are
// staged in LDS; rounds 2-3 only touch prefix-matching elements (L2-warm).
// Ties at the exact threshold are kept in ascending index order
// (deterministic, matches jax.lax.top_k).
// ---------------------------------------------------------------------------
__global__ __launch_bounds__(1024)
void topk_rowselect(float* __restrict__ F, int H, int Kkeep)
{
    const int row = blockIdx.x;
    const int tid = threadIdx.x;
    float* rowp = F + (size_t)row * H;

    __shared__ uint16_t sv16[16384];
    __shared__ uint32_t hist[256];
    __shared__ uint32_t s_sel[2];          // [0]=chosen bucket, [1]=remaining k
    __shared__ uint32_t warpCnt[32];
    __shared__ uint32_t warpOff[32];
    __shared__ uint32_t s_base;

    for (int i = tid; i < H; i += blockDim.x)
        sv16[i] = (uint16_t)(__float_as_uint(rowp[i]) >> 16);
    __syncthreads();

    uint32_t prefix = 0, maskSo = 0, k = (uint32_t)Kkeep;
    for (int round = 0; round < 4; ++round) {
        const int shift = 24 - 8 * round;
        if (tid < 256) hist[tid] = 0;
        __syncthreads();

        const uint16_t pref16 = (uint16_t)(prefix >> 16);
        for (int i = tid; i < H; i += blockDim.x) {
            const uint32_t hv = sv16[i];
            if (round == 0) {
                atomicAdd(&hist[hv >> 8], 1u);
            } else if (round == 1) {
                if ((hv >> 8) == (prefix >> 24))
                    atomicAdd(&hist[hv & 255u], 1u);
            } else {
                if (hv == pref16) {
                    const uint32_t v = __float_as_uint(rowp[i]);
                    if ((v & maskSo) == prefix)
                        atomicAdd(&hist[(v >> shift) & 255u], 1u);
                }
            }
        }
        __syncthreads();

        if (tid == 0) {
            uint32_t cum = 0;
            s_sel[0] = 0; s_sel[1] = 0;
            for (int b = 255; b >= 0; --b) {
                const uint32_t c = hist[b];
                if (cum + c >= k) { s_sel[0] = (uint32_t)b; s_sel[1] = k - cum; break; }
                cum += c;
            }
        }
        __syncthreads();
        prefix |= s_sel[0] << shift;
        k = s_sel[1];
        maskSo |= 0xFFu << shift;
        __syncthreads();
    }

    const uint32_t T     = prefix;          // exact K-th largest value (bits)
    const uint32_t kties = k;               // # of ==T elements to keep
    const uint16_t T16   = (uint16_t)(T >> 16);

    if (tid == 0) s_base = 0;
    __syncthreads();

    const int wid  = tid >> 5;
    const int lane = tid & 31;
    for (int chunk = 0; chunk < H; chunk += blockDim.x) {
        const int i = chunk + tid;
        const uint16_t hv = sv16[i];
        int cls;                            // 0 drop, 1 keep, 2 tie (==T)
        if (hv < T16)      cls = 0;
        else if (hv > T16) cls = 1;
        else {
            const uint32_t v = __float_as_uint(rowp[i]);
            cls = (v > T) ? 1 : ((v == T) ? 2 : 0);
        }
        const unsigned long long bal = __ballot(cls == 2);
        if (lane == 0) warpCnt[wid] = (uint32_t)__popcll(bal);
        __syncthreads();
        if (tid == 0) {
            uint32_t run = s_base;
            for (int w = 0; w < 32; ++w) { warpOff[w] = run; run += warpCnt[w]; }
            s_base = run;
        }
        __syncthreads();
        if (cls == 2) {
            const uint32_t rank = warpOff[wid] +
                (uint32_t)__popcll(bal & ((1ull << lane) - 1ull));
            if (rank >= kties) cls = 0;
        }
        if (cls == 0) rowp[i] = 0.0f;
        __syncthreads();
    }
}

// ---------------------------------------------------------------------------
// Launch: encoder GEMM -> in-place top-K -> decoder GEMM.
// d_out = [sparse_features (4096x16384) | reconstruction (4096x4096)] f32.
// No workspace needed; stream ordering provides inter-kernel dependencies.
// ---------------------------------------------------------------------------
extern "C" void kernel_launch(void* const* d_in, const int* in_sizes, int n_in,
                              void* d_out, int out_size, void* d_ws, size_t ws_size,
                              hipStream_t stream)
{
    (void)in_sizes; (void)n_in; (void)out_size; (void)d_ws; (void)ws_size;

    const float* x     = (const float*)d_in[0];   // [4,1024,4096]
    const float* W_enc = (const float*)d_in[1];   // [16384,4096]
    const float* b_enc = (const float*)d_in[2];   // [16384]
    const float* W_dec = (const float*)d_in[3];   // [4096,16384]

    constexpr int Mrows = 4 * 1024;   // B*S
    constexpr int Din   = 4096;
    constexpr int Hdim  = 16384;
    constexpr int Kkeep = 1024;

    float* sparse = (float*)d_out;                         // [Mrows, Hdim]
    float* recon  = (float*)d_out + (size_t)Mrows * Hdim;  // [Mrows, Din]

    // Encoder: features = relu(x @ W_enc^T + b_enc), written dense.
    gemm_nt_wmma<true><<<dim3(Hdim / 128, Mrows / 128), dim3(256), 0, stream>>>(
        x, W_enc, b_enc, sparse, Mrows, Hdim, Din);

    // Top-K per row, in place (zero everything below the K-th largest).
    topk_rowselect<<<dim3(Mrows), dim3(1024), 0, stream>>>(sparse, Hdim, Kkeep);

    // Decoder: reconstruction = sparse @ W_dec^T.
    gemm_nt_wmma<false><<<dim3(Din / 128, Mrows / 128), dim3(256), 0, stream>>>(
        sparse, W_dec, nullptr, recon, Mrows, Din, Hdim);
}
